// SCFlow_geo_57672820850961
// MI455X (gfx1250) — matrix operand order repaired and checked by reading, not compiled, not run on hardware
//
#include <hip/hip_runtime.h>
#include <hip/hip_fp16.h>

typedef __attribute__((ext_vector_type(16))) _Float16 v16h;
typedef __attribute__((ext_vector_type(8)))  float    v8f;

struct __align__(16) U4 { unsigned int x, y, z, w; };
union FragH { v16h h; U4 q[2]; };

// ---------------------------------------------------------------------------
// Generic implicit-GEMM conv: Y[m, n] = relu(bias[n] + sum_k A[m,k] * W[n,k])
//   A gathered on the fly (im2col) from NHWC f16 input x[B,H,W,Cin]
//   W pre-converted f16 [Cout, Kp], K-order = (kh, kw, ci), zero-padded
//   Output written f16 at y[m * ldY + cOff + n]  (enables free concat)
// Block: 256 threads (8 waves), tile 128(M) x 64(N), K-step 32.
// Each wave computes a 32x32 sub-tile with 4 x v_wmma_f32_16x16x32_f16.
// Division-free: Ho/Wo are powers of two (shifts), K decode is loop-carried.
// ---------------------------------------------------------------------------
__global__ __launch_bounds__(256) void conv_gemm_f16(
    const _Float16* __restrict__ x, const _Float16* __restrict__ wf,
    const float* __restrict__ bias, _Float16* __restrict__ y,
    int M, int H, int W, int Cin, int K, int Cout,
    int Kh, int Kw, int stride, int pad, int Kp,
    int woShift, int hoShift, int ldY, int cOff, int doRelu)
{
    __shared__ __align__(16) _Float16 As[128 * 40];  // 128 rows, stride 40 halves
    __shared__ __align__(16) _Float16 Bs[64 * 40];   // 64 cols(N), stride 40 halves

    const int tid  = threadIdx.x;
    const int lane = tid & 31;
    const int wave = tid >> 5;
    const int wm   = wave & 3;   // M sub-tile (32 rows each)
    const int wn   = wave >> 2;  // N sub-tile (32 cols each)
    const int hv   = lane >> 4;  // lane half (0/1)
    const int l    = lane & 15;

    const int m0 = blockIdx.x * 128;
    const int n0 = blockIdx.y * 64;

    // A staging: each thread owns one half-row of 16 halves (division-free decode)
    const int ar = tid >> 1;
    const int ak = (tid & 1) * 16;
    const int am = m0 + ar;
    const bool amOk = am < M;
    const int aw = am & ((1 << woShift) - 1);
    const int t0 = am >> woShift;
    const int ah = t0 & ((1 << hoShift) - 1);
    const int ab = t0 >> hoShift;
    const int hb = ah * stride - pad;
    const int wb = aw * stride - pad;
    const size_t rowBase = (size_t)ab * H * W;   // image base (in pixels)
    U4* const asDst = (U4*)&As[ar * 40 + ak];

    // loop-carried K decode of kk0 = k0 + ak  ->  (ci, kw, kh); no divisions
    int ci = ak, kw = 0, kh = 0;
    while (ci >= Cin) { ci -= Cin; if (++kw == Kw) { kw = 0; ++kh; } }

    // B staging: each thread owns one 8-half chunk of one weight row
    const int bn = tid >> 2;
    const int bk = (tid & 3) * 8;
    const bool bnOk = (n0 + bn) < Cout;

    const v8f vzero = {0.f, 0.f, 0.f, 0.f, 0.f, 0.f, 0.f, 0.f};
    v8f acc[2][2];
    acc[0][0] = vzero; acc[0][1] = vzero; acc[1][0] = vzero; acc[1][1] = vzero;

    for (int k0 = 0; k0 < Kp; k0 += 32) {
        // ---- stage A (implicit im2col gather) ----
        const int kk0 = k0 + ak;
        if (amOk && kk0 < K) {
            const int hi = hb + kh, wi = wb + kw;
            const size_t eoff = (rowBase + (size_t)hi * W + wi) * Cin + ci;
            if (ci + 16 <= Cin && kk0 + 16 <= K &&
                hi >= 0 && hi < H && wi >= 0 && wi < W && ((eoff & 7) == 0)) {
                // fast path: whole 16-half run contiguous + aligned -> 2x b128
                const U4* src = (const U4*)&x[eoff];
                asDst[0] = src[0];
                asDst[1] = src[1];
            } else {
                int c2 = ci, w2 = kw, h2 = kh;
                __align__(16) _Float16 tA[16];
                for (int i = 0; i < 16; ++i) {
                    float v = 0.0f;
                    if (kk0 + i < K) {
                        int hi2 = hb + h2, wi2 = wb + w2;
                        if (hi2 >= 0 && hi2 < H && wi2 >= 0 && wi2 < W)
                            v = (float)x[(rowBase + (size_t)hi2 * W + wi2) * Cin + c2];
                    }
                    tA[i] = (_Float16)v;
                    if (++c2 == Cin) { c2 = 0; if (++w2 == Kw) { w2 = 0; ++h2; } }
                }
                asDst[0] = *(const U4*)&tA[0];
                asDst[1] = *(const U4*)&tA[8];
            }
        } else {
            U4 z; z.x = z.y = z.z = z.w = 0u;
            asDst[0] = z;
            asDst[1] = z;
        }

        // ---- stage B (pre-converted f16 weights) ----
        U4 bv; bv.x = bv.y = bv.z = bv.w = 0u;
        if (bnOk) {
            const _Float16* src = wf + (size_t)(n0 + bn) * Kp + k0 + bk;
            bv = *(const U4*)src;
            __builtin_prefetch(src + 32, 0, 1);   // next K-step -> global_prefetch
        }
        *(U4*)&Bs[bn * 40 + bk] = bv;

        __syncthreads();

        // ---- fragment loads per documented wave32 WMMA layouts ----
        // A 16x32: lane l holds row M=l; VGPR pairs cover K = 8*hv..+7, 16+8*hv..+7
        // B 32x16: lane l holds col N=l; VGPR pairs cover K = 8*hv..+7, 16+8*hv..+7
        FragH fa0, fa1, fb0, fb1;
        const int ra0 = (wm * 32 + l) * 40;
        const int ra1 = (wm * 32 + 16 + l) * 40;
        const int cb0 = (wn * 32 + l) * 40;
        const int cb1 = (wn * 32 + 16 + l) * 40;
        fa0.q[0] = *(const U4*)&As[ra0 + 8 * hv];
        fa0.q[1] = *(const U4*)&As[ra0 + 16 + 8 * hv];
        fa1.q[0] = *(const U4*)&As[ra1 + 8 * hv];
        fa1.q[1] = *(const U4*)&As[ra1 + 16 + 8 * hv];
        fb0.q[0] = *(const U4*)&Bs[cb0 + 8 * hv];
        fb0.q[1] = *(const U4*)&Bs[cb0 + 16 + 8 * hv];
        fb1.q[0] = *(const U4*)&Bs[cb1 + 8 * hv];
        fb1.q[1] = *(const U4*)&Bs[cb1 + 16 + 8 * hv];

        acc[0][0] = __builtin_amdgcn_wmma_f32_16x16x32_f16(false, fa0.h, false, fb0.h, (short)0, acc[0][0], false, false);
        acc[0][1] = __builtin_amdgcn_wmma_f32_16x16x32_f16(false, fa0.h, false, fb1.h, (short)0, acc[0][1], false, false);
        acc[1][0] = __builtin_amdgcn_wmma_f32_16x16x32_f16(false, fa1.h, false, fb0.h, (short)0, acc[1][0], false, false);
        acc[1][1] = __builtin_amdgcn_wmma_f32_16x16x32_f16(false, fa1.h, false, fb1.h, (short)0, acc[1][1], false, false);

        __syncthreads();

        // advance loop-carried K decode by 32 (no divisions)
        ci += 32;
        while (ci >= Cin) { ci -= Cin; if (++kw == Kw) { kw = 0; ++kh; } }
    }

    // ---- epilogue: C/D layout (VGPR r: lanes0-15 -> M=r, lanes16-31 -> M=r+8; N=l)
    const int gn0 = n0 + wn * 32 + l;        // j = 0 column
    const int gn1 = gn0 + 16;                // j = 1 column
    const float bv0 = (bias && gn0 < Cout) ? bias[gn0] : 0.0f;
    const float bv1 = (bias && gn1 < Cout) ? bias[gn1] : 0.0f;
    for (int i = 0; i < 2; ++i)
        for (int j = 0; j < 2; ++j) {
            const int gn = j ? gn1 : gn0;
            const float bb = j ? bv1 : bv0;
            if (gn >= Cout) continue;
            #pragma unroll
            for (int r = 0; r < 8; ++r) {
                int gm = m0 + wm * 32 + i * 16 + r + 8 * hv;
                if (gm < M) {
                    float v = acc[i][j][r] + bb;
                    if (doRelu) v = fmaxf(v, 0.0f);
                    y[(size_t)gm * ldY + cOff + gn] = (_Float16)v;
                }
            }
        }
}

// ---------------------------------------------------------------------------
// NCHW f32 -> NHWC f16, division-free: grid = (pixels/256, C, B)
// ---------------------------------------------------------------------------
__global__ void nchw_to_nhwc_f16(const float* __restrict__ in, _Float16* __restrict__ out,
                                 int HW, int C, int ldY, int cOff)
{
    int hw = blockIdx.x * blockDim.x + threadIdx.x;
    if (hw >= HW) return;
    int c = blockIdx.y;
    int b = blockIdx.z;
    out[((size_t)b * HW + hw) * ldY + cOff + c] =
        (_Float16)in[((size_t)b * C + c) * HW + hw];
}

// ---------------------------------------------------------------------------
// Conv weight f32 [Co,Ci,Kh,Kw] -> f16 [Co, Kp], K-order (kh,kw,ci), zero-pad
// ---------------------------------------------------------------------------
__global__ void conv_w_to_f16(const float* __restrict__ w, _Float16* __restrict__ wf,
                              int Co, int Ci, int Kh, int Kw, int Kp)
{
    int i = blockIdx.x * blockDim.x + threadIdx.x;
    if (i >= Co * Kp) return;
    int co = i / Kp, k = i % Kp;
    float v = 0.0f;
    if (Kh == 1 && Kw == 1) {
        // FC / 1x1 fast path: identity K order, no divisions
        if (k < Ci) v = w[(size_t)co * Ci + k];
    } else if (k < Ci * Kh * Kw) {
        int ci = k % Ci; int t = k / Ci;
        int kw = t % Kw; int kh = t / Kw;
        v = w[(size_t)(((co * Ci + ci) * Kh + kh)) * Kw + kw];
    }
    wf[i] = (_Float16)v;
}

// ---------------------------------------------------------------------------
// GroupNorm(32, cpg=4) + ReLU, in-place on f16 NHWC. Block per (batch, group).
// ---------------------------------------------------------------------------
__global__ __launch_bounds__(256) void groupnorm_relu_f16(
    _Float16* __restrict__ x, const float* __restrict__ sc, const float* __restrict__ bi,
    int HW, int C)
{
    __shared__ float r1[256], r2[256];
    const int g = blockIdx.x & 31;          // 32 groups
    const int b = blockIdx.x >> 5;
    const size_t base = (size_t)b * HW * C + g * 4;
    const int N = HW * 4;                   // cpg = 4

    float s = 0.f, s2 = 0.f;
    for (int i = threadIdx.x; i < N; i += 256) {
        int hw = i >> 2, cc = i & 3;
        float v = (float)x[base + (size_t)hw * C + cc];
        s += v; s2 += v * v;
    }
    r1[threadIdx.x] = s; r2[threadIdx.x] = s2;
    __syncthreads();
    for (int st = 128; st > 0; st >>= 1) {
        if (threadIdx.x < st) { r1[threadIdx.x] += r1[threadIdx.x + st];
                                r2[threadIdx.x] += r2[threadIdx.x + st]; }
        __syncthreads();
    }
    const float mean = r1[0] / N;
    const float var  = r2[0] / N - mean * mean;
    const float rstd = rsqrtf(var + 1e-5f);
    const float sc0 = sc[g * 4 + 0], sc1 = sc[g * 4 + 1], sc2 = sc[g * 4 + 2], sc3 = sc[g * 4 + 3];
    const float bi0 = bi[g * 4 + 0], bi1 = bi[g * 4 + 1], bi2 = bi[g * 4 + 2], bi3 = bi[g * 4 + 3];
    for (int i = threadIdx.x; i < N; i += 256) {
        int hw = i >> 2, cc = i & 3;
        float scl = cc == 0 ? sc0 : cc == 1 ? sc1 : cc == 2 ? sc2 : sc3;
        float bia = cc == 0 ? bi0 : cc == 1 ? bi1 : cc == 2 ? bi2 : bi3;
        size_t idx = base + (size_t)hw * C + cc;
        float v = ((float)x[idx] - mean) * rstd * scl + bia;
        x[idx] = (_Float16)fmaxf(v, 0.0f);
    }
}

// ---------------------------------------------------------------------------
// Flatten [B,4,4,128] NHWC f16 -> [B,2048] f16 in reference (c,h,w) order
// ---------------------------------------------------------------------------
__global__ void pack_fc_kernel(const _Float16* __restrict__ in, _Float16* __restrict__ out, int total)
{
    int i = blockIdx.x * blockDim.x + threadIdx.x;
    if (i >= total) return;
    int b = i >> 11, k = i & 2047;
    int c = k >> 4,  r = k & 15;
    int h = r >> 2,  w = r & 3;
    out[i] = in[(size_t)((b * 4 + h) * 4 + w) * 128 + c];
}

// ---------------------------------------------------------------------------
// Heads: only columns selected by obj_id[0] are needed (9 outputs per batch)
// ---------------------------------------------------------------------------
__global__ void heads_kernel(const _Float16* __restrict__ x,
                             const float* __restrict__ rot_w, const float* __restrict__ rot_b,
                             const float* __restrict__ tr_w,  const float* __restrict__ tr_b,
                             const int* __restrict__ obj, float* __restrict__ out)
{
    int t = blockIdx.x * blockDim.x + threadIdx.x;
    if (t >= 64 * 9) return;
    int j = obj[0] - 1;
    if (t < 64 * 6) {
        int b = t / 6, e = t % 6;
        int row = j * 6 + e;
        const float* wr = rot_w + (size_t)row * 256;
        const _Float16* xb = x + (size_t)b * 256;
        float s = rot_b[row];
        for (int k = 0; k < 256; ++k) s += (float)xb[k] * wr[k];
        out[b * 6 + e] = s;
    } else {
        int t2 = t - 384;
        int b = t2 / 3, e = t2 % 3;
        int row = j * 3 + e;
        const float* wr = tr_w + (size_t)row * 256;
        const _Float16* xb = x + (size_t)b * 256;
        float s = tr_b[row];
        for (int k = 0; k < 256; ++k) s += (float)xb[k] * wr[k];
        out[384 + b * 3 + e] = s;
    }
}

// ---------------------------------------------------------------------------
extern "C" void kernel_launch(void* const* d_in, const int* in_sizes, int n_in,
                              void* d_out, int out_size, void* d_ws, size_t ws_size,
                              hipStream_t stream)
{
    (void)in_sizes; (void)n_in; (void)out_size; (void)ws_size;

    const float* hidden = (const float*)d_in[0];
    const float* dflow  = (const float*)d_in[1];
    const float* mask   = (const float*)d_in[2];
    const float* geo    = (const float*)d_in[3];
    const int*   obj    = (const int*)d_in[4];
    float* out = (float*)d_out;

    char* ws = (char*)d_ws;
    size_t off = 0;
    auto allocH = [&](size_t nHalf) -> _Float16* {
        _Float16* p = (_Float16*)(ws + off);
        off = (off + nHalf * sizeof(_Float16) + 255) & ~(size_t)255;
        return p;
    };

    // weights (f16, padded K)
    _Float16* wf_g1 = allocH(128 * 96);
    _Float16* wf_g2 = allocH(64 * 1152);
    _Float16* wf_d1 = allocH(128 * 128);
    _Float16* wf_d2 = allocH(64 * 1152);
    _Float16* wf_m1 = allocH(64 * 32);
    _Float16* wf_m2 = allocH(32 * 576);
    _Float16* wf_c1 = allocH(128 * 2592);
    _Float16* wf_c2 = allocH(128 * 1152);
    _Float16* wf_c3 = allocH(128 * 1152);
    _Float16* wf_f1 = allocH((size_t)1024 * 2048);
    _Float16* wf_f2 = allocH((size_t)256 * 1024);
    // activations (NHWC f16)
    _Float16* geoT = allocH((size_t)64 * 64 * 64 * 8);
    _Float16* dfT  = allocH((size_t)64 * 32 * 32 * 2);
    _Float16* mkT  = allocH((size_t)64 * 32 * 32 * 1);
    _Float16* g1o  = allocH((size_t)64 * 64 * 64 * 128);
    _Float16* d1o  = allocH((size_t)64 * 32 * 32 * 128);
    _Float16* m1o  = allocH((size_t)64 * 32 * 32 * 64);
    _Float16* enc  = allocH((size_t)64 * 32 * 32 * 288);
    _Float16* c1o  = allocH((size_t)64 * 16 * 16 * 128);
    _Float16* c2o  = allocH((size_t)64 * 8 * 8 * 128);
    _Float16* c3o  = allocH((size_t)64 * 4 * 4 * 128);
    _Float16* fcA  = allocH((size_t)64 * 2048);
    _Float16* f1o  = allocH((size_t)64 * 1024);
    _Float16* f2o  = allocH((size_t)64 * 256);

    // ---- weight conversion ----
    auto cw = [&](const void* w, _Float16* wf, int Co, int Ci, int Kh, int Kw, int Kp) {
        int total = Co * Kp;
        conv_w_to_f16<<<(total + 255) / 256, 256, 0, stream>>>((const float*)w, wf, Co, Ci, Kh, Kw, Kp);
    };
    cw(d_in[5],  wf_g1, 128,    8, 3, 3,   96);
    cw(d_in[7],  wf_g2,  64,  128, 3, 3, 1152);
    cw(d_in[9],  wf_d1, 128,    2, 7, 7,  128);
    cw(d_in[11], wf_d2,  64,  128, 3, 3, 1152);
    cw(d_in[13], wf_m1,  64,    1, 3, 3,   32);
    cw(d_in[15], wf_m2,  32,   64, 3, 3,  576);
    cw(d_in[17], wf_c1, 128,  288, 3, 3, 2592);
    cw(d_in[21], wf_c2, 128,  128, 3, 3, 1152);
    cw(d_in[25], wf_c3, 128,  128, 3, 3, 1152);
    cw(d_in[29], wf_f1, 1024, 2048, 1, 1, 2048);
    cw(d_in[31], wf_f2, 256,  1024, 1, 1, 1024);

    // ---- input layout: NCHW f32 -> NHWC f16 (hidden lands directly in enc) ----
    auto tr = [&](const float* in, _Float16* o, int B, int C, int H, int W, int ldY, int cOff) {
        int HW = H * W;
        dim3 grid((HW + 255) / 256, C, B);
        nchw_to_nhwc_f16<<<grid, 256, 0, stream>>>(in, o, HW, C, ldY, cOff);
    };
    tr(hidden, enc,  64, 128, 32, 32, 288, 0);
    tr(geo,    geoT, 64,   8, 64, 64,   8, 0);
    tr(dflow,  dfT,  64,   2, 32, 32,   2, 0);
    tr(mask,   mkT,  64,   1, 32, 32,   1, 0);

    // ---- convs / FCs via WMMA implicit GEMM ----
    auto conv = [&](const _Float16* x, const _Float16* w, const void* bias, _Float16* y,
                    int Bn, int H, int W, int Cin, int Ho, int Wo, int Cout,
                    int Kh, int Kw, int s, int p, int Kp,
                    int woShift, int hoShift, int ldY, int cOff, int relu) {
        int M = Bn * Ho * Wo;
        int K = Kh * Kw * Cin;
        dim3 grid((M + 127) / 128, (Cout + 63) / 64);
        conv_gemm_f16<<<grid, 256, 0, stream>>>(x, w, (const float*)bias, y,
            M, H, W, Cin, K, Cout, Kh, Kw, s, p, Kp, woShift, hoShift, ldY, cOff, relu);
    };

    // geo encoder
    conv(geoT, wf_g1, d_in[6], g1o, 64, 64, 64,   8, 64, 64, 128, 3, 3, 1, 1,   96, 6, 6, 128,   0, 1);
    conv(g1o,  wf_g2, d_in[8], enc, 64, 64, 64, 128, 32, 32,  64, 3, 3, 2, 1, 1152, 5, 5, 288, 224, 1);
    // delta-flow encoder
    conv(dfT,  wf_d1, d_in[10], d1o, 64, 32, 32,   2, 32, 32, 128, 7, 7, 1, 3,  128, 5, 5, 128,   0, 1);
    conv(d1o,  wf_d2, d_in[12], enc, 64, 32, 32, 128, 32, 32,  64, 3, 3, 1, 1, 1152, 5, 5, 288, 128, 1);
    // mask encoder
    conv(mkT,  wf_m1, d_in[14], m1o, 64, 32, 32,   1, 32, 32,  64, 3, 3, 1, 1,   32, 5, 5,  64,   0, 1);
    conv(m1o,  wf_m2, d_in[16], enc, 64, 32, 32,  64, 32, 32,  32, 3, 3, 1, 1,  576, 5, 5, 288, 192, 1);
    // conv layers (bias fused, ReLU after GN)
    conv(enc,  wf_c1, d_in[18], c1o, 64, 32, 32, 288, 16, 16, 128, 3, 3, 2, 1, 2592, 4, 4, 128, 0, 0);
    groupnorm_relu_f16<<<64 * 32, 256, 0, stream>>>(c1o, (const float*)d_in[19], (const float*)d_in[20], 256, 128);
    conv(c1o,  wf_c2, d_in[22], c2o, 64, 16, 16, 128,  8,  8, 128, 3, 3, 2, 1, 1152, 3, 3, 128, 0, 0);
    groupnorm_relu_f16<<<64 * 32, 256, 0, stream>>>(c2o, (const float*)d_in[23], (const float*)d_in[24],  64, 128);
    conv(c2o,  wf_c3, d_in[26], c3o, 64,  8,  8, 128,  4,  4, 128, 3, 3, 2, 1, 1152, 2, 2, 128, 0, 0);
    groupnorm_relu_f16<<<64 * 32, 256, 0, stream>>>(c3o, (const float*)d_in[27], (const float*)d_in[28],  16, 128);

    // FC layers (flatten in reference (c,h,w) order, then 1x1 "conv" GEMMs)
    pack_fc_kernel<<<(64 * 2048 + 255) / 256, 256, 0, stream>>>(c3o, fcA, 64 * 2048);
    conv(fcA, wf_f1, d_in[30], f1o, 64, 1, 1, 2048, 1, 1, 1024, 1, 1, 1, 0, 2048, 0, 0, 1024, 0, 1);
    conv(f1o, wf_f2, d_in[32], f2o, 64, 1, 1, 1024, 1, 1,  256, 1, 1, 1, 0, 1024, 0, 0,  256, 0, 1);

    // prediction heads + obj_id select -> d_out = [rot(64x6), tr(64x3)]
    heads_kernel<<<3, 192, 0, stream>>>(f2o,
        (const float*)d_in[33], (const float*)d_in[34],
        (const float*)d_in[35], (const float*)d_in[36], obj, out);
}
